// BezierParameterProcessor_7464653161142
// MI455X (gfx1250) — matrix operand order repaired and checked by reading, not compiled, not run on hardware
//
#include <hip/hip_runtime.h>

typedef __attribute__((ext_vector_type(16))) _Float16 v16h;
typedef __attribute__((ext_vector_type(8)))  _Float16 v8h;
typedef __attribute__((ext_vector_type(8)))  float    v8f;

#define D256 256
#define NB 8
#define NK 16
#define NR 100
#define NPTS (NK*NR)          // 1600
#define G 4096                // 64*64

// ---------------------------------------------------------------------------
// convert f32 weights [K][N] -> f16 fragment-major pack [K/32][N][hi(2)][16]
// so each lane's WMMA B-fragment is 16 contiguous halfs (two b128 loads).
// element (ks, n, hi, j)  <-  src[(ks*32 + hi*16 + j)*N + n]
// ---------------------------------------------------------------------------
__global__ void packB_kernel(const float* __restrict__ src, _Float16* __restrict__ dst,
                             int N, int K) {
    int i = blockIdx.x * blockDim.x + threadIdx.x;
    if (i >= K * N) return;
    int j    = i & 15;
    int hi   = (i >> 4) & 1;
    int rest = i >> 5;
    int n    = rest % N;
    int ks   = rest / N;
    int k    = ks * 32 + hi * 16 + j;
    dst[i] = (_Float16)src[(size_t)k * N + n];
}

// ---------------------------------------------------------------------------
// generic WMMA GEMM: one wave per 16x16 output tile.
// A:  [M,K] f16 row-major.   Wp: fragment-packed f16 (see packB_kernel).
// out = (A@W + bias), compile-time relu / f16-out / f32-out.
// ---------------------------------------------------------------------------
template<int RELU, int OUTH, int OUTF>
__global__ __launch_bounds__(32) void gemm_wmma(
    const _Float16* __restrict__ A, const _Float16* __restrict__ Wp,
    const float* __restrict__ bias,
    _Float16* __restrict__ outH, float* __restrict__ outF,
    int M, int N, int K)
{
    const int lane  = threadIdx.x;
    const int row16 = lane & 15;
    const int hi    = lane >> 4;            // 0: lanes 0-15, 1: lanes 16-31
    const int m0    = blockIdx.x * 16;
    const int n     = blockIdx.y * 16 + row16;

    v8f acc = {};
    const _Float16* Ar = A + (size_t)(m0 + row16) * K;
    const int ksteps = K >> 5;
    for (int ks = 0; ks < ksteps; ++ks) {
        const int ka = ks * 32 + (hi ? 16 : 0);
        v8h alo = *(const v8h*)(Ar + ka);
        v8h ahi = *(const v8h*)(Ar + ka + 8);
        v16h a = __builtin_shufflevector(alo, ahi, 0,1,2,3,4,5,6,7,8,9,10,11,12,13,14,15);
        v16h b = *(const v16h*)(Wp + ((size_t)((ks * N + n) * 2 + hi) << 4));
        acc = __builtin_amdgcn_wmma_f32_16x16x32_f16(false, a, false, b,
                                                     (short)0, acc, false, false);
    }
    const float bv = bias[n];
    #pragma unroll
    for (int v = 0; v < 8; ++v) {
        const int m = m0 + v + hi * 8;
        float val = acc[v] + bv;
        if (RELU) val = fmaxf(val, 0.f);
        const size_t o = (size_t)m * N + n;
        if (OUTH) outH[o] = (_Float16)val;
        if (OUTF) outF[o] = val;
    }
}

// ---------------------------------------------------------------------------
// encoder layer 1: [512,2] @ [2,64] + b, relu -> f16  (K=2: VALU, not WMMA)
// ---------------------------------------------------------------------------
__global__ void enc_l1_kernel(const float* __restrict__ cp, const float* __restrict__ w1,
                              const float* __restrict__ b1, _Float16* __restrict__ out) {
    int i = blockIdx.x * blockDim.x + threadIdx.x;   // 512*64
    int r = i >> 6, n = i & 63;
    float v = cp[r * 2] * w1[n] + cp[r * 2 + 1] * w1[64 + n] + b1[n];
    out[i] = (_Float16)fmaxf(v, 0.f);
}

// mean over 4 control points: h3f[512,256] -> cf_h[128,256]
__global__ void mean4_kernel(const float* __restrict__ h3, _Float16* __restrict__ cf) {
    int i = blockIdx.x * blockDim.x + threadIdx.x;   // 128*256
    int row = i >> 8, d = i & 255;
    float v = 0.25f * (h3[(row * 4 + 0) * D256 + d] + h3[(row * 4 + 1) * D256 + d] +
                       h3[(row * 4 + 2) * D256 + d] + h3[(row * 4 + 3) * D256 + d]);
    cf[i] = (_Float16)v;
}

// bn1 + bn2 + per-scale multiply -> f_h[3][128][256] f16
__global__ void make_f_kernel(const float* __restrict__ cf2, const float* __restrict__ g1,
                              const float* __restrict__ b1, const float* __restrict__ g2,
                              const float* __restrict__ b2, _Float16* __restrict__ fh) {
    int i = blockIdx.x * blockDim.x + threadIdx.x;   // 3*128*256
    int d = i & 255, row = (i >> 8) & 127, s = i >> 15;
    float sc  = (s == 0) ? 0.5f : ((s == 1) ? 1.0f : 2.0f);
    float inv = rsqrtf(1.0f + 1e-5f);
    float v = cf2[row * D256 + d] * (g1[d] * inv) + b1[d];
    v = v * (g2[d] * inv) + b2[d];
    fh[i] = (_Float16)(v * sc);
}

// kernel-size head: kh2[384,32] . w3 + b3 -> softplus -> 1/(2*ksz^2)
__global__ void ker_final_kernel(const _Float16* __restrict__ kh2, const float* __restrict__ w3,
                                 const float* __restrict__ b3, float* __restrict__ inv2s2) {
    int i = blockIdx.x * blockDim.x + threadIdx.x;
    if (i >= 384) return;
    float acc = b3[0];
    #pragma unroll
    for (int j = 0; j < 32; ++j) acc += (float)kh2[i * 32 + j] * w3[j];
    float sp = log1pf(__expf(acc));                  // softplus
    inv2s2[i] = 1.0f / (2.0f * sp * sp + 1e-20f);
}

// ---------------------------------------------------------------------------
// Bezier evaluation + per-batch min/max normalization -> pn[8][1600][2]
// ---------------------------------------------------------------------------
__global__ __launch_bounds__(256) void bezier_kernel(const float* __restrict__ cp,
                                                     float* __restrict__ pn) {
    const int b = blockIdx.x, t = threadIdx.x;
    __shared__ float sm[4][256];
    float mnx = 1e30f, mny = 1e30f, mxx = -1e30f, mxy = -1e30f;
    for (int n = t; n < NPTS; n += 256) {
        int k = n / NR, r = n % NR;
        float tt = r * (1.0f / (NR - 1));
        float u = 1.0f - tt;
        float c0 = u * u * u, c1 = 3.f * tt * u * u, c2 = 3.f * tt * tt * u, c3 = tt * tt * tt;
        const float* c = cp + (size_t)((b * NK + k) * 4) * 2;
        float x = c0 * c[0] + c1 * c[2] + c2 * c[4] + c3 * c[6];
        float y = c0 * c[1] + c1 * c[3] + c2 * c[5] + c3 * c[7];
        pn[(b * NPTS + n) * 2]     = x;
        pn[(b * NPTS + n) * 2 + 1] = y;
        mnx = fminf(mnx, x); mxx = fmaxf(mxx, x);
        mny = fminf(mny, y); mxy = fmaxf(mxy, y);
    }
    sm[0][t] = mnx; sm[1][t] = mny; sm[2][t] = mxx; sm[3][t] = mxy;
    __syncthreads();
    for (int s = 128; s > 0; s >>= 1) {
        if (t < s) {
            sm[0][t] = fminf(sm[0][t], sm[0][t + s]);
            sm[1][t] = fminf(sm[1][t], sm[1][t + s]);
            sm[2][t] = fmaxf(sm[2][t], sm[2][t + s]);
            sm[3][t] = fmaxf(sm[3][t], sm[3][t + s]);
        }
        __syncthreads();
    }
    const float ox = sm[0][0], oy = sm[1][0];
    const float sx = 1.0f / (sm[2][0] - ox + 1e-8f);
    const float sy = 1.0f / (sm[3][0] - oy + 1e-8f);
    for (int n = t; n < NPTS; n += 256) {
        float x = pn[(b * NPTS + n) * 2], y = pn[(b * NPTS + n) * 2 + 1];
        pn[(b * NPTS + n) * 2]     = (x - ox) * sx;
        pn[(b * NPTS + n) * 2 + 1] = (y - oy) * sy;
    }
}

// ---------------------------------------------------------------------------
// fused attention MLP: h1 = relu(base1 + pn-rank2) in LDS,
// h2 = relu(h1@W2+b2) via WMMA (W2 fragment-packed, direct from L2),
// attn = sigmoid(h2.w3 + b3).  64 points per 128-thread block.
// ---------------------------------------------------------------------------
__global__ __launch_bounds__(128) void attn_fused_kernel(
    const float* __restrict__ base1,   // [384][256] rows = (s*8+b)*16+k, includes b1
    const float* __restrict__ pn,      // [8][1600][2]
    const float* __restrict__ w1p,     // att_w1 rows 256..257 : [2][256]
    const _Float16* __restrict__ W2p,  // fragment-packed [256/32][128][2][16]
    const float* __restrict__ b2,      // [128]
    const float* __restrict__ w3,      // [128]
    const float* __restrict__ b3,      // [1]
    float* __restrict__ attn)          // [3][8][1600] flat
{
    __shared__ __align__(16) unsigned char buf[64 * 256 * sizeof(_Float16)]; // 32KB reused
    __shared__ float rowPx[64], rowPy[64];
    __shared__ int   rowBase[64];
    _Float16* h1 = (_Float16*)buf;     // [64][256] f16
    float*    h2 = (float*)buf;        // [64][128] f32 (after h1 is dead)

    const int t  = threadIdx.x;
    const int p0 = blockIdx.x * 64;

    if (t < 64) {
        int p   = p0 + t;
        int s8b = p / NPTS;
        int n   = p - s8b * NPTS;
        int b   = s8b & 7;
        int k   = n / NR;
        rowBase[t] = (s8b * NK + k) * D256;
        rowPx[t]   = pn[(b * NPTS + n) * 2];
        rowPy[t]   = pn[(b * NPTS + n) * 2 + 1];
    }
    __syncthreads();

    // phase 1: build h1 (relu of base1 + rank-2 pn term)
    for (int idx = t; idx < 64 * D256; idx += 128) {
        int r = idx >> 8, d = idx & 255;
        float v = base1[rowBase[r] + d] + rowPx[r] * w1p[d] + rowPy[r] * w1p[D256 + d];
        h1[idx] = (_Float16)fmaxf(v, 0.f);
    }
    __syncthreads();

    // phase 2: h2 = h1 @ W2 via WMMA; wave w owns rows w*16..w*16+15, cols 0..127
    const int lane  = t & 31;
    const int wave  = t >> 5;
    const int row16 = lane & 15;
    const int hi    = lane >> 4;
    const int r0    = wave * 16;
    v8f zero = {};
    v8f acc[8];
    #pragma unroll
    for (int i = 0; i < 8; ++i) acc[i] = zero;

    for (int ks = 0; ks < 8; ++ks) {
        const int ka = ks * 32 + (hi ? 16 : 0);
        const _Float16* hr = h1 + (r0 + row16) * D256;
        v8h alo = *(const v8h*)(hr + ka);
        v8h ahi = *(const v8h*)(hr + ka + 8);
        v16h a = __builtin_shufflevector(alo, ahi, 0,1,2,3,4,5,6,7,8,9,10,11,12,13,14,15);
        #pragma unroll
        for (int nt = 0; nt < 8; ++nt) {
            const int n = nt * 16 + row16;
            v16h b = *(const v16h*)(W2p + ((size_t)((ks * 128 + n) * 2 + hi) << 4));
            acc[nt] = __builtin_amdgcn_wmma_f32_16x16x32_f16(false, a, false, b,
                                                             (short)0, acc[nt], false, false);
        }
    }
    __syncthreads();   // all h1 reads done before overwriting with h2

    // phase 3: relu + bias, spill h2 to LDS (reuses h1 space)
    #pragma unroll
    for (int nt = 0; nt < 8; ++nt) {
        const int n = nt * 16 + row16;
        const float bv = b2[n];
        #pragma unroll
        for (int v = 0; v < 8; ++v) {
            const int m = r0 + v + hi * 8;
            h2[m * 128 + n] = fmaxf(acc[nt][v] + bv, 0.f);
        }
    }
    __syncthreads();

    // phase 4: dot with w3, sigmoid, store
    if (t < 64) {
        float dot = b3[0];
        for (int j = 0; j < 128; ++j) dot += h2[t * 128 + j] * w3[j];
        attn[p0 + t] = 1.0f / (1.0f + __expf(-dot));
    }
}

// ---------------------------------------------------------------------------
// KDE: one thread per pixel, 3 scales fused (d2 computed once), LDS-staged pts
// ---------------------------------------------------------------------------
__global__ __launch_bounds__(256) void kde_kernel(
    const float* __restrict__ pn, const float* __restrict__ attn,
    const float* __restrict__ inv2s2, float* __restrict__ maps) {
    __shared__ float sx[256], sy[256], sa0[256], sa1[256], sa2[256],
                     si0[256], si1[256], si2[256];
    const int g   = blockIdx.x * 256 + threadIdx.x;   // 0..32767
    const int b   = g >> 12;
    const int pix = g & 4095;
    const float gx = (pix & 63) * (1.f / 63.f);
    const float gy = (pix >> 6) * (1.f / 63.f);
    float a0 = 0.f, a1 = 0.f, a2 = 0.f;
    for (int n0 = 0; n0 < NPTS; n0 += 256) {
        const int t = threadIdx.x, n = n0 + t;
        if (n < NPTS) {
            sx[t] = pn[(b * NPTS + n) * 2];
            sy[t] = pn[(b * NPTS + n) * 2 + 1];
            const int k = n / NR;
            sa0[t] = attn[(0 * NB + b) * NPTS + n]; si0[t] = inv2s2[(0 * NB + b) * NK + k];
            sa1[t] = attn[(1 * NB + b) * NPTS + n]; si1[t] = inv2s2[(1 * NB + b) * NK + k];
            sa2[t] = attn[(2 * NB + b) * NPTS + n]; si2[t] = inv2s2[(2 * NB + b) * NK + k];
        }
        __syncthreads();
        const int cnt = min(256, NPTS - n0);
        for (int j = 0; j < cnt; ++j) {
            const float dx = sx[j] - gx, dy = sy[j] - gy;
            const float d2 = dx * dx + dy * dy;
            a0 += sa0[j] * __expf(-d2 * si0[j]);
            a1 += sa1[j] * __expf(-d2 * si1[j]);
            a2 += sa2[j] * __expf(-d2 * si2[j]);
        }
        __syncthreads();
    }
    maps[(b * 3 + 0) * G + pix] = a0;
    maps[(b * 3 + 1) * G + pix] = a1;
    maps[(b * 3 + 2) * G + pix] = a2;
}

// ---------------------------------------------------------------------------
// conv fusion: 3x3(3->16) relu, 3x3(16->8) relu, 1x1(8->1) sigmoid, bn3
// ---------------------------------------------------------------------------
__global__ __launch_bounds__(256) void fuse_kernel(
    const float* __restrict__ maps,
    const float* __restrict__ w1, const float* __restrict__ b1,
    const float* __restrict__ w2, const float* __restrict__ b2,
    const float* __restrict__ w3, const float* __restrict__ b3,
    const float* __restrict__ g3, const float* __restrict__ bb3,
    float* __restrict__ out) {
    __shared__ float sw1[16 * 27], sb1[16], sw2[8 * 144], sb2[8], sw3[8];
    {
        const int t = threadIdx.x;
        for (int i = t; i < 16 * 27; i += 256) sw1[i] = w1[i];
        for (int i = t; i < 8 * 144; i += 256) sw2[i] = w2[i];
        if (t < 16) sb1[t] = b1[t];
        if (t < 8)  { sb2[t] = b2[t]; sw3[t] = w3[t]; }
    }
    __syncthreads();
    const int g   = blockIdx.x * 256 + threadIdx.x;
    const int b   = g >> 12;
    const int pix = g & 4095;
    const int y = pix >> 6, x = pix & 63;
    const float* mb = maps + (size_t)b * 3 * G;

    float acc2[8];
    #pragma unroll
    for (int o = 0; o < 8; ++o) acc2[o] = sb2[o];

    for (int dy = -1; dy <= 1; ++dy) {
        for (int dx = -1; dx <= 1; ++dx) {
            const int yy = y + dy, xx = x + dx;
            if (yy < 0 || yy >= 64 || xx < 0 || xx >= 64) continue;
            float c1[16];
            #pragma unroll
            for (int o = 0; o < 16; ++o) c1[o] = sb1[o];
            for (int ky = -1; ky <= 1; ++ky) {
                const int sy2 = yy + ky;
                if (sy2 < 0 || sy2 >= 64) continue;
                for (int kx = -1; kx <= 1; ++kx) {
                    const int sx2 = xx + kx;
                    if (sx2 < 0 || sx2 >= 64) continue;
                    const int wi = (ky + 1) * 3 + (kx + 1);
                    #pragma unroll
                    for (int ic = 0; ic < 3; ++ic) {
                        const float m = mb[ic * G + sy2 * 64 + sx2];
                        #pragma unroll
                        for (int o = 0; o < 16; ++o)
                            c1[o] += m * sw1[o * 27 + ic * 9 + wi];
                    }
                }
            }
            const int w2i = (dy + 1) * 3 + (dx + 1);
            #pragma unroll
            for (int ic = 0; ic < 16; ++ic) {
                const float v = fmaxf(c1[ic], 0.f);
                #pragma unroll
                for (int o = 0; o < 8; ++o)
                    acc2[o] += v * sw2[o * 144 + ic * 9 + w2i];
            }
        }
    }
    float v3 = b3[0];
    #pragma unroll
    for (int o = 0; o < 8; ++o) v3 += fmaxf(acc2[o], 0.f) * sw3[o];
    const float sg = 1.0f / (1.0f + __expf(-v3));
    out[g] = sg * (g3[0] * rsqrtf(1.0f + 1e-5f)) + bb3[0];
}

// ---------------------------------------------------------------------------
extern "C" void kernel_launch(void* const* d_in, const int* in_sizes, int n_in,
                              void* d_out, int out_size, void* d_ws, size_t ws_size,
                              hipStream_t stream) {
    const float* cp     = (const float*)d_in[0];
    const float* enc_w1 = (const float*)d_in[1];
    const float* enc_b1 = (const float*)d_in[2];
    const float* enc_w2 = (const float*)d_in[3];
    const float* enc_b2 = (const float*)d_in[4];
    const float* enc_w3 = (const float*)d_in[5];
    const float* enc_b3 = (const float*)d_in[6];
    const float* agg_w1 = (const float*)d_in[7];
    const float* agg_b1 = (const float*)d_in[8];
    const float* agg_w2 = (const float*)d_in[9];
    const float* agg_b2 = (const float*)d_in[10];
    const float* bn1_g  = (const float*)d_in[11];
    const float* bn1_b  = (const float*)d_in[12];
    const float* bn2_g  = (const float*)d_in[13];
    const float* bn2_b  = (const float*)d_in[14];
    const float* att_w1 = (const float*)d_in[15];
    const float* att_b1 = (const float*)d_in[16];
    const float* att_w2 = (const float*)d_in[17];
    const float* att_b2 = (const float*)d_in[18];
    const float* att_w3 = (const float*)d_in[19];
    const float* att_b3 = (const float*)d_in[20];
    const float* ker_w1 = (const float*)d_in[21];
    const float* ker_b1 = (const float*)d_in[22];
    const float* ker_w2 = (const float*)d_in[23];
    const float* ker_b2 = (const float*)d_in[24];
    const float* ker_w3 = (const float*)d_in[25];
    const float* ker_b3 = (const float*)d_in[26];
    const float* fus_w1 = (const float*)d_in[27];
    const float* fus_b1 = (const float*)d_in[28];
    const float* fus_w2 = (const float*)d_in[29];
    const float* fus_b2 = (const float*)d_in[30];
    const float* fus_w3 = (const float*)d_in[31];
    const float* fus_b3 = (const float*)d_in[32];
    const float* bn3_g  = (const float*)d_in[33];
    const float* bn3_b  = (const float*)d_in[34];
    float* out = (float*)d_out;

    // ---- workspace layout
    size_t off = 0;
    auto alloc = [&](size_t bytes) -> char* {
        char* p = (char*)d_ws + off;
        off += (bytes + 255) & ~(size_t)255;
        return p;
    };
    _Float16* wp_enc_w2 = (_Float16*)alloc(64 * 128 * 2);
    _Float16* wp_enc_w3 = (_Float16*)alloc(128 * 256 * 2);
    _Float16* wp_agg_w1 = (_Float16*)alloc(256 * 256 * 2);
    _Float16* wp_agg_w2 = (_Float16*)alloc(256 * 256 * 2);
    _Float16* wp_att_w1 = (_Float16*)alloc(256 * 256 * 2);   // feature rows only
    _Float16* wp_att_w2 = (_Float16*)alloc(256 * 128 * 2);
    _Float16* wp_ker_w1 = (_Float16*)alloc(256 * 64 * 2);
    _Float16* wp_ker_w2 = (_Float16*)alloc(64 * 32 * 2);
    _Float16* h1e   = (_Float16*)alloc(512 * 64 * 2);
    _Float16* h2e   = (_Float16*)alloc(512 * 128 * 2);
    float*    h3f   = (float*)   alloc(512 * 256 * 4);
    _Float16* cf_h  = (_Float16*)alloc(128 * 256 * 2);
    _Float16* a1_h  = (_Float16*)alloc(128 * 256 * 2);
    float*    cf2f  = (float*)   alloc(128 * 256 * 4);
    _Float16* f_h   = (_Float16*)alloc(3 * 128 * 256 * 2);
    float*    base1 = (float*)   alloc(3 * 128 * 256 * 4);
    _Float16* kh1   = (_Float16*)alloc(384 * 64 * 2);
    _Float16* kh2   = (_Float16*)alloc(384 * 32 * 2);
    float*    inv2  = (float*)   alloc(3 * 128 * 4);
    float*    pn    = (float*)   alloc(NB * NPTS * 2 * 4);
    float*    attn  = (float*)   alloc(3 * NB * NPTS * 4);
    float*    maps  = (float*)   alloc(NB * 3 * G * 4);
    (void)ws_size; (void)n_in; (void)in_sizes; (void)out_size;

    // ---- weight conversion + fragment packing (f32 -> f16)
    auto pack = [&](const float* s, _Float16* d, int N, int K) {
        int n = K * N;
        packB_kernel<<<(n + 255) / 256, 256, 0, stream>>>(s, d, N, K);
    };
    pack(enc_w2, wp_enc_w2, 128, 64);
    pack(enc_w3, wp_enc_w3, 256, 128);
    pack(agg_w1, wp_agg_w1, 256, 256);
    pack(agg_w2, wp_agg_w2, 256, 256);
    pack(att_w1, wp_att_w1, 256, 256);   // rows 0..255 (feature part)
    pack(att_w2, wp_att_w2, 128, 256);
    pack(ker_w1, wp_ker_w1, 64, 256);
    pack(ker_w2, wp_ker_w2, 32, 64);

    // ---- per-control-point encoder
    enc_l1_kernel<<<(512 * 64) / 256, 256, 0, stream>>>(cp, enc_w1, enc_b1, h1e);
    gemm_wmma<1,1,0><<<dim3(512 / 16, 128 / 16), 32, 0, stream>>>(
        h1e, wp_enc_w2, enc_b2, h2e, nullptr, 512, 128, 64);
    gemm_wmma<1,0,1><<<dim3(512 / 16, 256 / 16), 32, 0, stream>>>(
        h2e, wp_enc_w3, enc_b3, nullptr, h3f, 512, 256, 128);
    mean4_kernel<<<(128 * 256) / 256, 256, 0, stream>>>(h3f, cf_h);
    gemm_wmma<1,1,0><<<dim3(128 / 16, 256 / 16), 32, 0, stream>>>(
        cf_h, wp_agg_w1, agg_b1, a1_h, nullptr, 128, 256, 256);
    gemm_wmma<1,0,1><<<dim3(128 / 16, 256 / 16), 32, 0, stream>>>(
        a1_h, wp_agg_w2, agg_b2, nullptr, cf2f, 128, 256, 256);

    // ---- bn1 + bn2 + scale -> f (only 384 distinct feature rows)
    make_f_kernel<<<(3 * 128 * 256) / 256, 256, 0, stream>>>(
        cf2f, bn1_g, bn1_b, bn2_g, bn2_b, f_h);

    // ---- attention base (feature part of layer 1, bias folded, no relu yet)
    gemm_wmma<0,0,1><<<dim3(384 / 16, 256 / 16), 32, 0, stream>>>(
        f_h, wp_att_w1, att_b1, nullptr, base1, 384, 256, 256);

    // ---- kernel-size MLP (384 rows)
    gemm_wmma<1,1,0><<<dim3(384 / 16, 64 / 16), 32, 0, stream>>>(
        f_h, wp_ker_w1, ker_b1, kh1, nullptr, 384, 64, 256);
    gemm_wmma<1,1,0><<<dim3(384 / 16, 32 / 16), 32, 0, stream>>>(
        kh1, wp_ker_w2, ker_b2, kh2, nullptr, 384, 32, 64);
    ker_final_kernel<<<3, 128, 0, stream>>>(kh2, ker_w3, ker_b3, inv2);

    // ---- Bezier points + normalize
    bezier_kernel<<<NB, 256, 0, stream>>>(cp, pn);

    // ---- fused attention MLP (WMMA layer 2)
    attn_fused_kernel<<<(3 * NB * NPTS) / 64, 128, 0, stream>>>(
        base1, pn, att_w1 + 256 * 256, wp_att_w2, att_b2, att_w3, att_b3, attn);

    // ---- KDE density maps (3 scales fused)
    kde_kernel<<<(NB * G) / 256, 256, 0, stream>>>(pn, attn, inv2, maps);

    // ---- conv fusion + bn3
    fuse_kernel<<<(NB * G) / 256, 256, 0, stream>>>(
        maps, fus_w1, fus_b1, fus_w2, fus_b2, fus_w3, fus_b3, bn3_g, bn3_b, out);
}
